// CharRNNAttention_18184891531995
// MI455X (gfx1250) — matrix operand order repaired and checked by reading, not compiled, not run on hardware
//
#include <hip/hip_runtime.h>

typedef __attribute__((ext_vector_type(16))) _Float16 v16h;
typedef __attribute__((ext_vector_type(8)))  float    v8f;

#define Bsz 32
#define Tln 256
#define Vocab 256
#define Dd 64
#define Hh 256
#define Aa 128
#define NB 16   // cooperating workgroups in the recurrence phase

#define WMMA(a,b,c) __builtin_amdgcn_wmma_f32_16x16x32_f16(false,(a),false,(b),(short)0,(c),false,false)

__device__ __forceinline__ float fast_tanh(float v) {
#if __has_builtin(__builtin_amdgcn_tanhf)
  return __builtin_amdgcn_tanhf(v);         // v_tanh_f32 (CDNA5 TRANS op)
#elif __has_builtin(__builtin_amdgcn_tanh_f32)
  return __builtin_amdgcn_tanh_f32(v);
#else
  return tanhf(v);
#endif
}

// ---------- fragment index helpers ----------
// B-fragment (32x16 f16 tile): lane = nn | (kk[3]<<4), e = (kk[4]<<3)|kk[2:0]
__device__ __forceinline__ int bfragIdx(int k, int n, int KT) {
  int kt = k >> 5, kk = k & 31, nt = n >> 4, nn = n & 15;
  return ((nt * KT + kt) << 9) + (nn | (((kk >> 3) & 1) << 4)) * 16 + ((((kk >> 4) & 1) << 3) | (kk & 7));
}
// A-fragment (16x32 f16 tile): lane = mm | (kk[3]<<4), e = (kk[4]<<3)|kk[2:0]
__device__ __forceinline__ int afragLE(int mm, int kk) {
  int ln = (mm & 15) | (((kk >> 3) & 1) << 4);
  int e  = (((kk >> 4) & 1) << 3) | (kk & 7);
  return ln * 16 + e;
}

// ---------- weight packing ----------
__global__ void pack_wcat(const float* __restrict__ Wih, const float* __restrict__ Whh,
                          _Float16* __restrict__ out) {
  for (int i = blockIdx.x * blockDim.x + threadIdx.x; i < 576 * 256; i += gridDim.x * blockDim.x) {
    int k = i >> 8, n = i & 255;
    float v = (k < 320) ? Wih[k * 256 + n] : Whh[(k - 320) * 256 + n];
    out[bfragIdx(k, n, 18)] = (_Float16)v;
  }
}
__global__ void pack_wqk(const float* __restrict__ Wq, const float* __restrict__ Wk,
                         _Float16* __restrict__ out) {
  for (int i = blockIdx.x * blockDim.x + threadIdx.x; i < 256 * 256; i += gridDim.x * blockDim.x) {
    int k = i >> 8, n = i & 255;
    float v = (n < 128) ? Wq[k * 128 + n] : Wk[k * 128 + (n - 128)];
    out[bfragIdx(k, n, 8)] = (_Float16)v;
  }
}
__global__ void pack_wout(const float* __restrict__ Wo, _Float16* __restrict__ out) {
  for (int i = blockIdx.x * blockDim.x + threadIdx.x; i < 512 * 256; i += gridDim.x * blockDim.x) {
    int k = i >> 8, n = i & 255;
    out[bfragIdx(k, n, 16)] = (_Float16)Wo[k * 256 + n];
  }
}

__global__ void init_sync(unsigned* s) { if (threadIdx.x < 64) s[threadIdx.x] = 0u; }

// ---------- inter-workgroup generation barrier (agent scope) ----------
__device__ __forceinline__ void gridBarrier(unsigned* cnt, unsigned* gen, unsigned nb) {
  __threadfence();                       // release: all stores visible at device scope
  __syncthreads();
  if (threadIdx.x == 0) {
    unsigned g = __hip_atomic_load(gen, __ATOMIC_RELAXED, __HIP_MEMORY_SCOPE_AGENT);
    if (__hip_atomic_fetch_add(cnt, 1u, __ATOMIC_ACQ_REL, __HIP_MEMORY_SCOPE_AGENT) == nb - 1u) {
      __hip_atomic_store(cnt, 0u, __ATOMIC_RELAXED, __HIP_MEMORY_SCOPE_AGENT);
      __hip_atomic_store(gen, g + 1u, __ATOMIC_RELEASE, __HIP_MEMORY_SCOPE_AGENT);
    } else {
      while (__hip_atomic_load(gen, __ATOMIC_ACQUIRE, __HIP_MEMORY_SCOPE_AGENT) == g)
        __builtin_amdgcn_s_sleep(2);
    }
  }
  __syncthreads();
  __builtin_amdgcn_fence(__ATOMIC_ACQUIRE, "agent");  // invalidate WGP-level caches
}

// ---------- cooperative recurrence: block 0 = GEMMs, all 16 blocks = attention ----------
// q/keys layout: float4-packed [a4][b] -> float index ((a>>2)*32 + b)*4 + (a&3)
__global__ __launch_bounds__(512) void rnn_recurrence(
    const int* __restrict__ x, const float* __restrict__ E,
    const float* __restrict__ b_ih, const float* __restrict__ v_a,
    const _Float16* __restrict__ WcatF, const _Float16* __restrict__ WqkF,
    float* __restrict__ hidbuf,   // [t][b][j]  T*B*H f32
    float* __restrict__ keysbuf,  // [t][a4][b][4]  T*A*B f32
    float* __restrict__ qbuf,     // [a4][b][4]     A*B f32
    float* __restrict__ mbuf,     // [w][b]     NB*B f32
    float* __restrict__ Sbuf,     // [w][b]     NB*B f32
    float* __restrict__ ctxp,     // [w][b][j]  NB*B*H f32
    _Float16* __restrict__ hcF,   // A-frags of (8192 x 512) [h|ctx]
    float* __restrict__ out_h,    // d_out + B*T*V
    unsigned* __restrict__ sync)
{
  extern __shared__ char smem[];
  _Float16* aF     = (_Float16*)smem;             // 18432 halves
  float*    ctxL   = (float*)(smem + 36864);      // 8192
  float*    vaL    = (float*)(smem + 69632);      // 128
  float*    bL     = (float*)(smem + 70144);      // 256
  float*    qcache = (float*)(smem + 71168);      // 4096, [a4][b][4]
  float*    sliceSc= (float*)(smem + 87552);      // 512
  float*    fwL    = (float*)(smem + 89600);      // 512
  float*    SinvL  = (float*)(smem + 91648);      // 32

  const int tid  = threadIdx.x;
  const int lane = tid & 31;
  const int wave = tid >> 5;
  const int blk  = blockIdx.x;

  if (tid < 128) vaL[tid] = v_a[tid];
  if (blk == 0) {
    if (tid < 256) bL[tid] = b_ih[tid];
    for (int i = tid; i < 8192; i += 512)  ctxL[i] = 0.0f;
    for (int i = tid; i < 18432; i += 512) aF[i] = (_Float16)0.0f;  // h_prev = 0
  }
  __syncthreads();

  for (int t = 0; t < Tln; ++t) {
    if (blk == 0) {
      // ---- combine partial softmax/context of step t-1 ----
      if (t > 0) {
        if (tid < 32) {
          float M = -1e30f;
          for (int w = 0; w < NB; ++w) M = fmaxf(M, mbuf[w * 32 + tid]);
          float S = 0.0f;
          for (int w = 0; w < NB; ++w) {
            float f = __expf(mbuf[w * 32 + tid] - M);
            fwL[w * 32 + tid] = f;
            S += Sbuf[w * 32 + tid] * f;
          }
          SinvL[tid] = 1.0f / S;
        }
        __syncthreads();
        int gmtBase = ((t - 1) * 32) >> 4;
        for (int it = tid; it < 2048; it += 512) {
          int b = it >> 6, j4 = it & 63;
          float4 cv = make_float4(0.f, 0.f, 0.f, 0.f);
          for (int w = 0; w < NB; ++w) {
            float f = fwL[w * 32 + b];
            float4 pv = *((const float4*)ctxp + ((size_t)w * 32 + b) * 64 + j4);
            cv.x += f * pv.x; cv.y += f * pv.y; cv.z += f * pv.z; cv.w += f * pv.w;
          }
          float si = SinvL[b];
          cv.x *= si; cv.y *= si; cv.z *= si; cv.w *= si;
          ((float4*)ctxL)[it] = cv;
          int j = j4 * 4;
          int base = ((gmtBase + (b >> 4)) * 16 + (8 + (j >> 5))) << 9;
          hcF[base + afragLE(b & 15, (j + 0) & 31)] = (_Float16)cv.x;
          hcF[base + afragLE(b & 15, (j + 1) & 31)] = (_Float16)cv.y;
          hcF[base + afragLE(b & 15, (j + 2) & 31)] = (_Float16)cv.z;
          hcF[base + afragLE(b & 15, (j + 3) & 31)] = (_Float16)cv.w;
        }
        __syncthreads();
      }

      // ---- build A-frags kt 0..9 ([e_t | ctx], 32 x 320) ----
      for (int idx = tid; idx < 32 * 320; idx += 512) {
        int m = idx / 320, k = idx - m * 320;
        float v = (k < 64) ? E[x[m * Tln + t] * Dd + k] : ctxL[m * 256 + (k - 64)];
        aF[(((m >> 4) * 18 + (k >> 5)) << 9) + afragLE(m & 15, k & 31)] = (_Float16)v;
      }
      __syncthreads();

      // ---- GEMM1: (32 x 576) @ (576 x 256), both m-tiles share nt = wave ----
      const _Float16* wcB = WcatF + ((wave * 18) << 9) + lane * 16;
      v8f c0 = {}, c1 = {};
      v16h b0  = *(const v16h*)(wcB);
      v16h a00 = *(const v16h*)(aF + lane * 16);
      v16h a10 = *(const v16h*)(aF + (18 << 9) + lane * 16);
#pragma unroll 2
      for (int kt = 0; kt < 18; ++kt) {
        v16h bn = b0, an0 = a00, an1 = a10;
        if (kt < 17) {
          bn  = *(const v16h*)(wcB + ((kt + 1) << 9));
          an0 = *(const v16h*)(aF + ((kt + 1) << 9) + lane * 16);
          an1 = *(const v16h*)(aF + ((18 + kt + 1) << 9) + lane * 16);
        }
        c0 = WMMA(a00, b0, c0);
        c1 = WMMA(a10, b0, c1);
        b0 = bn; a00 = an0; a10 = an1;
      }
      __syncthreads();   // everyone done reading h_prev frags

      // ---- epilogue: h = tanh(pre + b_ih) ----
      {
        int n = wave * 16 + (lane & 15);
        float bias = bL[n];
        int kslot = (10 + (n >> 5)) << 9;
        int hslot = n >> 5;
        for (int i = 0; i < 2; ++i) {
          v8f c = i ? c1 : c0;
          for (int r = 0; r < 8; ++r) {
            int m = i * 16 + ((lane >> 4) * 8 + r);
            float hv = fast_tanh(c[r] + bias);
            int le = afragLE(m & 15, n & 31);
            hidbuf[(t * 32 + m) * 256 + n] = hv;
            aF[((i * 18) << 9) + kslot + le] = (_Float16)hv;
            int gmt = (t * 32 + m) >> 4;
            hcF[((gmt * 16 + hslot) << 9) + le] = (_Float16)hv;
          }
        }
      }
      __syncthreads();

      // ---- GEMM-qk: (32 x 256) @ (256 x 256) -> [q | keys_t] ----
      const _Float16* wqB = WqkF + ((wave * 8) << 9) + lane * 16;
      v8f d0 = {}, d1 = {};
      v16h e0  = *(const v16h*)(wqB);
      v16h f00 = *(const v16h*)(aF + (10 << 9) + lane * 16);
      v16h f10 = *(const v16h*)(aF + ((18 + 10) << 9) + lane * 16);
#pragma unroll 2
      for (int kt = 0; kt < 8; ++kt) {
        v16h en = e0, fn0 = f00, fn1 = f10;
        if (kt < 7) {
          en  = *(const v16h*)(wqB + ((kt + 1) << 9));
          fn0 = *(const v16h*)(aF + ((10 + kt + 1) << 9) + lane * 16);
          fn1 = *(const v16h*)(aF + ((18 + 10 + kt + 1) << 9) + lane * 16);
        }
        d0 = WMMA(f00, e0, d0);
        d1 = WMMA(f10, e0, d1);
        e0 = en; f00 = fn0; f10 = fn1;
      }
      {
        int n = wave * 16 + (lane & 15);
        for (int i = 0; i < 2; ++i) {
          v8f d = i ? d1 : d0;
          for (int r = 0; r < 8; ++r) {
            int m = i * 16 + ((lane >> 4) * 8 + r);
            if (n < 128) qbuf[(n >> 2) * 128 + m * 4 + (n & 3)] = d[r];
            else {
              int a = n - 128;
              keysbuf[(size_t)t * 4096 + (a >> 2) * 128 + m * 4 + (a & 3)] = d[r];
            }
          }
        }
      }
    }

    gridBarrier(sync, sync + 1, NB);   // h_t, q_t, keys_t visible device-wide

    // ---- all blocks: partial scores / softmax / context over s-slice ----
    for (int i = tid; i < 1024; i += 512)
      ((float4*)qcache)[i] = ((const float4*)qbuf)[i];
    __syncthreads();

    int nls = (blk <= t) ? ((t - blk) / NB + 1) : 0;
    {
      int ls = tid >> 5, b = tid & 31;
      float sc = -1e30f;
      if (ls < nls) {
        int s = blk + ls * NB;
        const float4* kp = (const float4*)keysbuf + (size_t)s * 1024 + b;
        const float4* qp = (const float4*)qcache + b;
        const float4* vp = (const float4*)vaL;
        float acc = 0.0f;
        for (int a4 = 0; a4 < 32; ++a4) {
          float4 kv = kp[a4 * 32];
          float4 qv = qp[a4 * 32];
          float4 vv = vp[a4];
          acc += fast_tanh(qv.x + kv.x) * vv.x;
          acc += fast_tanh(qv.y + kv.y) * vv.y;
          acc += fast_tanh(qv.z + kv.z) * vv.z;
          acc += fast_tanh(qv.w + kv.w) * vv.w;
        }
        sc = acc;
      }
      sliceSc[ls * 32 + b] = sc;
    }
    __syncthreads();
    if (tid < 32) {
      float mx = -1e30f;
      for (int ls = 0; ls < NB; ++ls) mx = fmaxf(mx, sliceSc[ls * 32 + tid]);
      float sum = 0.0f;
      for (int ls = 0; ls < NB; ++ls) {
        float v = sliceSc[ls * 32 + tid];
        float e = (v > -1e29f) ? __expf(v - mx) : 0.0f;
        sliceSc[ls * 32 + tid] = e;
        sum += e;
      }
      mbuf[blk * 32 + tid] = mx;
      Sbuf[blk * 32 + tid] = sum;
    }
    __syncthreads();
    for (int it = tid; it < 2048; it += 512) {
      int b = it >> 6, j4 = it & 63;
      float4 cv = make_float4(0.f, 0.f, 0.f, 0.f);
      for (int ls = 0; ls < nls; ++ls) {
        float e = sliceSc[ls * 32 + b];
        float4 hv = *((const float4*)hidbuf + ((size_t)(blk + ls * NB) * 32 + b) * 64 + j4);
        cv.x += e * hv.x; cv.y += e * hv.y; cv.z += e * hv.z; cv.w += e * hv.w;
      }
      *((float4*)ctxp + ((size_t)blk * 32 + b) * 64 + j4) = cv;
    }

    gridBarrier(sync, sync + 1, NB);   // partials visible device-wide
  }

  // ---- block 0: final combine (t=255) + final hidden state ----
  if (blk == 0) {
    if (tid < 32) {
      float M = -1e30f;
      for (int w = 0; w < NB; ++w) M = fmaxf(M, mbuf[w * 32 + tid]);
      float S = 0.0f;
      for (int w = 0; w < NB; ++w) {
        float f = __expf(mbuf[w * 32 + tid] - M);
        fwL[w * 32 + tid] = f;
        S += Sbuf[w * 32 + tid] * f;
      }
      SinvL[tid] = 1.0f / S;
    }
    __syncthreads();
    for (int it = tid; it < 2048; it += 512) {
      int b = it >> 6, j4 = it & 63;
      float4 cv = make_float4(0.f, 0.f, 0.f, 0.f);
      for (int w = 0; w < NB; ++w) {
        float f = fwL[w * 32 + b];
        float4 pv = *((const float4*)ctxp + ((size_t)w * 32 + b) * 64 + j4);
        cv.x += f * pv.x; cv.y += f * pv.y; cv.z += f * pv.z; cv.w += f * pv.w;
      }
      float si = SinvL[b];
      cv.x *= si; cv.y *= si; cv.z *= si; cv.w *= si;
      int j = j4 * 4;
      int base = (((255 * 32 + b) >> 4) * 16 + (8 + (j >> 5))) << 9;
      hcF[base + afragLE(b & 15, (j + 0) & 31)] = (_Float16)cv.x;
      hcF[base + afragLE(b & 15, (j + 1) & 31)] = (_Float16)cv.y;
      hcF[base + afragLE(b & 15, (j + 2) & 31)] = (_Float16)cv.z;
      hcF[base + afragLE(b & 15, (j + 3) & 31)] = (_Float16)cv.w;
      ((float4*)out_h)[it] = *((const float4*)hidbuf + ((size_t)(255 * 32 + b)) * 64 + j4);
    }
  }
}

// ---------- output projection: (8192 x 512) @ (512 x 256) + b_out ----------
__global__ __launch_bounds__(256) void out_gemm(
    const _Float16* __restrict__ hcF, const _Float16* __restrict__ WoutF,
    const float* __restrict__ b_out, float* __restrict__ out)
{
  int lane = threadIdx.x & 31, wave = threadIdx.x >> 5;
  int tileId = blockIdx.x * 8 + wave;       // 0..8191
  int mt = tileId >> 4, nt = tileId & 15;
  const _Float16* aB = hcF + ((mt * 16) << 9) + lane * 16;
  const _Float16* bB = WoutF + ((nt * 16) << 9) + lane * 16;
  v8f c = {};
  v16h a0 = *(const v16h*)aB;
  v16h b0 = *(const v16h*)bB;
#pragma unroll 2
  for (int kt = 0; kt < 16; ++kt) {
    v16h an = a0, bn = b0;
    if (kt < 15) {
      an = *(const v16h*)(aB + ((kt + 1) << 9));
      bn = *(const v16h*)(bB + ((kt + 1) << 9));
    }
    c = WMMA(a0, b0, c);
    a0 = an; b0 = bn;
  }
  int n = nt * 16 + (lane & 15);
  float bias = b_out[n];
  for (int r = 0; r < 8; ++r) {
    int m  = mt * 16 + ((lane >> 4) * 8 + r);   // m = t*32 + b
    int b_ = m & 31, t_ = m >> 5;
    out[((size_t)b_ * 256 + t_) * 256 + n] = c[r] + bias;   // (B,T,V)
  }
}

extern "C" void kernel_launch(void* const* d_in, const int* in_sizes, int n_in,
                              void* d_out, int out_size, void* d_ws, size_t ws_size,
                              hipStream_t stream) {
  const int*   x     = (const int*)  d_in[0];
  const float* E     = (const float*)d_in[1];
  const float* W_ih  = (const float*)d_in[2];
  const float* b_ih  = (const float*)d_in[3];
  const float* W_hh  = (const float*)d_in[4];
  const float* W_q   = (const float*)d_in[5];
  const float* W_k   = (const float*)d_in[6];
  const float* v_a   = (const float*)d_in[7];
  const float* W_out = (const float*)d_in[8];
  const float* b_out = (const float*)d_in[9];
  float* out = (float*)d_out;

  char* ws = (char*)d_ws;
  _Float16* WcatF   = (_Float16*)(ws);              // 294912
  _Float16* WqkF    = (_Float16*)(ws + 294912);     // 131072
  _Float16* WoutF   = (_Float16*)(ws + 425984);     // 262144
  _Float16* hcF     = (_Float16*)(ws + 688128);     // 8388608
  float*    keysbuf = (float*)   (ws + 9076736);    // 4194304
  float*    hidbuf  = (float*)   (ws + 13271040);   // 8388608
  float*    qbuf    = (float*)   (ws + 21659648);   // 16384
  float*    mbuf    = (float*)   (ws + 21676032);   // 2048
  float*    Sbuf    = (float*)   (ws + 21678080);   // 2048
  float*    ctxp    = (float*)   (ws + 21680128);   // 524288
  unsigned* syncv   = (unsigned*)(ws + 22204416);   // 256
  // total 22204672 bytes (~21.2 MB)

  init_sync<<<1, 64, 0, stream>>>(syncv);
  pack_wcat<<<144, 256, 0, stream>>>(W_ih, W_hh, WcatF);
  pack_wqk <<<64,  256, 0, stream>>>(W_q, W_k, WqkF);
  pack_wout<<<128, 256, 0, stream>>>(W_out, WoutF);

  const size_t lds = 91904;
  rnn_recurrence<<<NB, 512, lds, stream>>>(x, E, b_ih, v_a, WcatF, WqkF,
                                           hidbuf, keysbuf, qbuf, mbuf, Sbuf, ctxp,
                                           hcF, out + (size_t)Bsz * Tln * Vocab, syncv);

  out_gemm<<<1024, 256, 0, stream>>>(hcF, WoutF, b_out, out);
}
